// Matformer_81157702025409
// MI455X (gfx1250) — matrix operand order repaired and checked by reading, not compiled, not run on hardware
//
#include <hip/hip_runtime.h>
#include <hip/hip_bf16.h>
#include <math.h>

// ---------------------------------------------------------------------------
// Types for CDNA5 WMMA
// ---------------------------------------------------------------------------
typedef _Float16 h16;
typedef __attribute__((ext_vector_type(8)))  _Float16 v8h;
typedef __attribute__((ext_vector_type(16))) _Float16 v16h;
typedef __attribute__((ext_vector_type(8)))  float    v8f;
typedef int v4i_ __attribute__((vector_size(16)));   // matches async-LDS builtin param type

#define TILE_LDA 40   // 32 halfs + 8 pad => 80B rows, keeps 16B alignment, kills bank conflicts

// Async direct global->LDS path (CDNA5 GLOBAL_LOAD_ASYNC_TO_LDS_B128, ASYNCcnt).
// Guarded so the kernel still compiles (with the sync path) if the
// toolchain doesn't declare the builtins.
#if defined(__has_builtin)
#  if __has_builtin(__builtin_amdgcn_global_load_async_to_lds_b128)
#    define HAVE_ASYNC_LDS 1
#  else
#    define HAVE_ASYNC_LDS 0
#  endif
#  if __has_builtin(__builtin_amdgcn_s_wait_asynccnt)
#    define HAVE_ASYNC_WAIT_BUILTIN 1
#  else
#    define HAVE_ASYNC_WAIT_BUILTIN 0
#  endif
#else
#  define HAVE_ASYNC_LDS 0
#  define HAVE_ASYNC_WAIT_BUILTIN 0
#endif

#define AS1 __attribute__((address_space(1)))
#define AS3 __attribute__((address_space(3)))

__device__ __forceinline__ void async_wait0()
{
#if HAVE_ASYNC_LDS
#  if HAVE_ASYNC_WAIT_BUILTIN
    __builtin_amdgcn_s_wait_asynccnt(0);
#  else
    asm volatile("s_wait_asynccnt 0x0" ::: "memory");
#  endif
#endif
}

// ---------------------------------------------------------------------------
// Tiled f16 WMMA GEMM:  C[M,N] = act(A[M,Kp] @ Bt[N,Kp]^T + bias) (* Gmul)
//   - A row-major f16, Bt is B pre-transposed [N,Kp] f16 (weights)
//   - 256 threads = 8 wave32, block tile 128x128, wave tile 64x32, K-step 32
//   - double-buffered LDS; async global->LDS on full tiles when available
//   - ACT: 0 none, 1 softplus ; O32/O16 select f32 and/or f16 outputs
//   - MUL: elementwise multiply by Gmul[M,N] (f16) in epilogue (gated message)
// Requires: Kp % 32 == 0, N % 128 == 0 (all our GEMMs satisfy this).
// ---------------------------------------------------------------------------
template<int ACT, bool O32, bool O16, bool MUL>
__global__ __launch_bounds__(256)
void gemm_f16_kernel(const h16* __restrict__ A, const h16* __restrict__ Bt,
                     const float* __restrict__ bias, const h16* __restrict__ Gmul,
                     float* __restrict__ C32, h16* __restrict__ C16,
                     int M, int N, int Kp)
{
    __shared__ h16 lA[2][128 * TILE_LDA];
    __shared__ h16 lB[2][128 * TILE_LDA];

    const int tid  = threadIdx.x;
    const int lane = tid & 31;
    const int wave = tid >> 5;
    const int wm   = wave >> 2;       // 0..1 : 64-row slab
    const int wn   = wave & 3;        // 0..3 : 32-col slab
    const int m0   = blockIdx.x * 128;
    const int n0   = blockIdx.y * 128;
    const int l16  = lane & 15;
    const int lhi  = lane >> 4;       // 0/1

    const int row0 = tid >> 2;        // 0..63
    const int seg0 = tid & 3;         // 16B segment within a 64B K-row

    const bool fullTile = (m0 + 128 <= M);   // N tiles are always full (N%128==0)

    v8f acc[4][2];
#pragma unroll
    for (int r = 0; r < 4; ++r)
#pragma unroll
        for (int c = 0; c < 2; ++c)
#pragma unroll
            for (int i = 0; i < 8; ++i) acc[r][c][i] = 0.0f;

    // ---- tile loader: async direct-to-LDS when the tile is fully in range ----
    auto load_tile = [&](int buf, int k0) {
#if HAVE_ASYNC_LDS
        if (fullTile) {
#pragma unroll
            for (int half = 0; half < 2; ++half) {
                const int row = row0 + half * 64;
                __builtin_amdgcn_global_load_async_to_lds_b128(
                    (AS1 v4i_*)(A + (size_t)(m0 + row) * Kp + k0 + seg0 * 8),
                    (AS3 v4i_*)&lA[buf][row * TILE_LDA + seg0 * 8], 0, 0);
                __builtin_amdgcn_global_load_async_to_lds_b128(
                    (AS1 v4i_*)(Bt + (size_t)(n0 + row) * Kp + k0 + seg0 * 8),
                    (AS3 v4i_*)&lB[buf][row * TILE_LDA + seg0 * 8], 0, 0);
            }
            return;
        }
#endif
#pragma unroll
        for (int half = 0; half < 2; ++half) {
            const int row = row0 + half * 64;
            v8h av; v8h bv;
#pragma unroll
            for (int i = 0; i < 8; ++i) { av[i] = (h16)0.0f; bv[i] = (h16)0.0f; }
            const int gm = m0 + row;
            if (gm < M) av = *(const v8h*)(A  + (size_t)gm * Kp + k0 + seg0 * 8);
            const int gn = n0 + row;
            if (gn < N) bv = *(const v8h*)(Bt + (size_t)gn * Kp + k0 + seg0 * 8);
            *(v8h*)&lA[buf][row * TILE_LDA + seg0 * 8] = av;
            *(v8h*)&lB[buf][row * TILE_LDA + seg0 * 8] = bv;
        }
    };

    // ---- double-buffered main loop ----
    load_tile(0, 0);
    async_wait0();
    __syncthreads();

    int buf = 0;
    for (int k0 = 0; k0 < Kp; k0 += 32) {
        const bool more = (k0 + 32 < Kp);
        if (more) load_tile(buf ^ 1, k0 + 32);

        // LDS -> fragments (ISA 16-bit A 16x32 / B 32x16 lane layouts)
        v16h af[4], bf[2];
#pragma unroll
        for (int r = 0; r < 4; ++r) {
            const int m = wm * 64 + r * 16 + l16;
            const int g = lhi * 8;                 // K group {0..7,16..23} / {8..15,24..31}
            v8h lo = *(const v8h*)&lA[buf][m * TILE_LDA + g];
            v8h hi = *(const v8h*)&lA[buf][m * TILE_LDA + g + 16];
#pragma unroll
            for (int i = 0; i < 8; ++i) { af[r][i] = lo[i]; af[r][8 + i] = hi[i]; }
        }
#pragma unroll
        for (int c = 0; c < 2; ++c) {
            const int n  = wn * 32 + c * 16 + l16;
            const int kb = lhi * 8;                // K group {0..7,16..23} / {8..15,24..31}
            v8h lo = *(const v8h*)&lB[buf][n * TILE_LDA + kb];
            v8h hi = *(const v8h*)&lB[buf][n * TILE_LDA + kb + 16];
#pragma unroll
            for (int i = 0; i < 8; ++i) { bf[c][i] = lo[i]; bf[c][8 + i] = hi[i]; }
        }

#pragma unroll
        for (int r = 0; r < 4; ++r)
#pragma unroll
            for (int c = 0; c < 2; ++c)
                acc[r][c] = __builtin_amdgcn_wmma_f32_16x16x32_f16(
                    false, af[r], false, bf[c], (short)0, acc[r][c], false, false);

        if (more) async_wait0();   // next tile landed in LDS (own wave's loads)
        __syncthreads();           // all waves' loads visible; safe to consume/overwrite
        buf ^= 1;
    }

    // ---- epilogue: bias / activation / gate-multiply / store ----
#pragma unroll
    for (int c = 0; c < 2; ++c) {
        const int nn = n0 + wn * 32 + c * 16 + l16;
        const float bvv = bias ? bias[nn] : 0.0f;
#pragma unroll
        for (int r = 0; r < 4; ++r) {
#pragma unroll
            for (int vr = 0; vr < 8; ++vr) {
                const int mm = m0 + wm * 64 + r * 16 + lhi * 8 + vr;
                if (mm < M) {
                    float v = acc[r][c][vr] + bvv;
                    if (ACT == 1) v = (v > 20.0f) ? v : log1pf(expf(v));
                    if (MUL) v *= (float)Gmul[(size_t)mm * N + nn];
                    if (O32) C32[(size_t)mm * N + nn] = v;
                    if (O16) C16[(size_t)mm * N + nn] = (h16)v;
                }
            }
        }
    }
}

// ---------------------------------------------------------------------------
// Small elementwise / prep kernels
// ---------------------------------------------------------------------------
__global__ void wtr_kernel(const float* __restrict__ W, h16* __restrict__ Wt,
                           int K, int N2, int Kp)
{   // W[K,N2] f32 -> Wt[N2,Kp] f16 (zero-padded K)
    int i = blockIdx.x * blockDim.x + threadIdx.x;
    if (i >= N2 * Kp) return;
    int n = i / Kp, k = i % Kp;
    Wt[i] = (k < K) ? (h16)W[(size_t)k * N2 + n] : (h16)0.0f;
}

__global__ void cvt_x_kernel(const float* __restrict__ x, h16* __restrict__ xh, int Nn)
{   // [Nn,92] f32 -> [Nn,96] f16 zero-padded
    int i = blockIdx.x * blockDim.x + threadIdx.x;
    if (i >= Nn * 96) return;
    int n = i / 96, k = i % 96;
    xh[i] = (k < 92) ? (h16)x[(size_t)n * 92 + k] : (h16)0.0f;
}

__global__ void cvt_f32_f16_kernel(const float* __restrict__ a, h16* __restrict__ b, long long n)
{
    long long i = (long long)blockIdx.x * blockDim.x + threadIdx.x;
    if (i < n) b[i] = (h16)a[i];
}

__global__ void fill0_kernel(float* __restrict__ p, long long n)
{
    long long i = (long long)blockIdx.x * blockDim.x + threadIdx.x;
    if (i < n) p[i] = 0.0f;
}

__global__ __launch_bounds__(128)
void rbf_kernel(const float* __restrict__ ea, h16* __restrict__ rbfh, int E_)
{
    int e = blockIdx.x, c = threadIdx.x;
    float a0 = ea[(size_t)e * 3 + 0], a1 = ea[(size_t)e * 3 + 1], a2 = ea[(size_t)e * 3 + 2];
    float d = sqrtf(a0 * a0 + a1 * a1 + a2 * a2);
    float center = (float)c * (8.0f / 127.0f);
    float gamma  = 127.0f / 8.0f;
    float t = d - center;
    rbfh[(size_t)e * 128 + c] = (h16)expf(-gamma * t * t);
}

// Build alpha = q3*k3*inv_sqrt, gate = sigmoid(LN(alpha)), m3 = [v_i|v_j|e]
__global__ __launch_bounds__(384)
void edge_prep_kernel(const int* __restrict__ eidx, int E_, int e0,
                      const h16* __restrict__ qh, const h16* __restrict__ kh,
                      const h16* __restrict__ vh, const h16* __restrict__ ehc,
                      const float* __restrict__ ln1g, const float* __restrict__ ln1b,
                      h16* __restrict__ gate, h16* __restrict__ m3h, float inv_sqrt)
{
    const int row = blockIdx.x;            // (edge-in-chunk, head)
    const int ec = row >> 2, h = row & 3;
    const int e = e0 + ec;
    const int src = eidx[e], dst = eidx[E_ + e];
    const int t = threadIdx.x;             // 0..383
    const int seg = t >> 7, c = t & 127;

    const size_t bi = (size_t)dst * 512 + h * 128 + c;
    float q_i = (float)qh[bi];
    float kv, mv;
    if (seg == 0)      { kv = (float)kh[bi]; mv = (float)vh[bi]; }
    else if (seg == 1) { const size_t bj = (size_t)src * 512 + h * 128 + c;
                         kv = (float)kh[bj]; mv = (float)vh[bj]; }
    else               { const size_t bo = (size_t)ec * 512 + h * 128 + c;
                         float ev = (float)ehc[bo]; kv = ev; mv = ev; }
    float alpha = q_i * kv * inv_sqrt;

    __shared__ float ssum[12], ssq[12];
    float s1 = alpha, s2 = alpha * alpha;
    for (int o = 16; o; o >>= 1) { s1 += __shfl_xor(s1, o, 32); s2 += __shfl_xor(s2, o, 32); }
    if ((t & 31) == 0) { ssum[t >> 5] = s1; ssq[t >> 5] = s2; }
    __syncthreads();
    float S = 0.0f, Q = 0.0f;
#pragma unroll
    for (int i = 0; i < 12; ++i) { S += ssum[i]; Q += ssq[i]; }
    float mu  = S * (1.0f / 384.0f);
    float var = Q * (1.0f / 384.0f) - mu * mu;
    float rstd = rsqrtf(var + 1e-5f);
    float gn = (alpha - mu) * rstd * ln1g[t] + ln1b[t];
    float gv = 1.0f / (1.0f + expf(-gn));

    const size_t o = (size_t)row * 384 + t;
    gate[o] = (h16)gv;
    m3h[o]  = (h16)mv;
}

// LayerNorm(C=128) on messages, then segment-sum into agg[dst,H*C] via atomics
__global__ __launch_bounds__(128)
void msg_scatter_kernel(const float* __restrict__ msg, const int* __restrict__ eidx,
                        int E_, int e0, const float* __restrict__ ln2g,
                        const float* __restrict__ ln2b, float* __restrict__ agg)
{
    const int row = blockIdx.x;
    const int ec = row >> 2, h = row & 3;
    const int dst = eidx[E_ + e0 + ec];
    const int c = threadIdx.x;
    float v = msg[(size_t)row * 128 + c];

    __shared__ float ssum[4], ssq[4];
    float s1 = v, s2 = v * v;
    for (int off = 16; off; off >>= 1) { s1 += __shfl_xor(s1, off, 32); s2 += __shfl_xor(s2, off, 32); }
    if ((c & 31) == 0) { ssum[c >> 5] = s1; ssq[c >> 5] = s2; }
    __syncthreads();
    float S = ssum[0] + ssum[1] + ssum[2] + ssum[3];
    float Q = ssq[0] + ssq[1] + ssq[2] + ssq[3];
    float mu  = S * (1.0f / 128.0f);
    float var = Q * (1.0f / 128.0f) - mu * mu;
    float rstd = rsqrtf(var + 1e-5f);
    float o = (v - mu) * rstd * ln2g[c] + ln2b[c];
    atomicAdd(&agg[(size_t)dst * 512 + h * 128 + c], o);
}

// nf = SiLU(nf + eval-BN(out)); re-emit f32 + f16 node features
__global__ __launch_bounds__(128)
void residual_kernel(const float* __restrict__ ob, const float* __restrict__ bng,
                     const float* __restrict__ bnb, float* __restrict__ nf,
                     h16* __restrict__ nfh)
{
    const int n = blockIdx.x, c = threadIdx.x;
    const size_t i = (size_t)n * 128 + c;
    float o = ob[i] * 0.9999950000374997f * bng[c] + bnb[c];   // 1/sqrt(1+1e-5)
    float xv = nf[i] + o;
    float s = xv / (1.0f + expf(-xv));
    nf[i]  = s;
    nfh[i] = (h16)s;
}

__global__ __launch_bounds__(128)
void pool_sum_kernel(const float* __restrict__ nf, const int* __restrict__ batch,
                     float* __restrict__ pooled, float* __restrict__ cnts)
{
    const int n = blockIdx.x, c = threadIdx.x;
    const int g = batch[n];
    atomicAdd(&pooled[(size_t)g * 128 + c], nf[(size_t)n * 128 + c]);
    if (c == 0) atomicAdd(&cnts[g], 1.0f);
}

// scatter-mean finalize + fc(SiLU) + out head; one block per graph
__global__ __launch_bounds__(128)
void head_kernel(const float* __restrict__ pooled, const float* __restrict__ cnts,
                 const float* __restrict__ fcW, const float* __restrict__ fcb,
                 const float* __restrict__ outW, const float* __restrict__ outb,
                 float* __restrict__ out)
{
    const int g = blockIdx.x, c = threadIdx.x;
    __shared__ float p[128];
    __shared__ float f[128];
    float cnt = fmaxf(cnts[g], 1.0f);
    p[c] = pooled[(size_t)g * 128 + c] / cnt;
    __syncthreads();
    float acc = fcb[c];
    for (int k = 0; k < 128; ++k) acc += p[k] * fcW[(size_t)k * 128 + c];
    float s = acc / (1.0f + expf(-acc));
    f[c] = s * outW[c];
    __syncthreads();
    for (int off = 64; off; off >>= 1) { if (c < off) f[c] += f[c + off]; __syncthreads(); }
    if (c == 0) out[g] = f[0] + outb[0];
}

// ---------------------------------------------------------------------------
// Host orchestration
// ---------------------------------------------------------------------------
#define GEMM(ACT, O32, O16, MUL, A_, Bt_, bias_, G_, C32_, C16_, M_, N_, K_)              \
    gemm_f16_kernel<ACT, O32, O16, MUL><<<dim3(((M_) + 127) / 128, (N_) / 128), 256, 0,   \
        stream>>>(A_, Bt_, bias_, G_, C32_, C16_, M_, N_, K_)

extern "C" void kernel_launch(void* const* d_in, const int* in_sizes, int n_in,
                              void* d_out, int out_size, void* d_ws, size_t ws_size,
                              hipStream_t stream)
{
    const int Nn = in_sizes[0] / 92;
    const int E_ = in_sizes[1] / 3;
    const int G_ = out_size;
    const int H_ = 4, NL_ = 5;
    const int CH = 8192;                       // edge chunk -> intermediates stay in L2

    const float* x      = (const float*)d_in[0];
    const float* eattr  = (const float*)d_in[1];
    const int*   eidx   = (const int*)d_in[2];
    const int*   batch  = (const int*)d_in[3];
    const float* emb_W  = (const float*)d_in[4];
    const float* emb_b  = (const float*)d_in[5];
    const float* rbf_W1 = (const float*)d_in[6];
    const float* rbf_b1 = (const float*)d_in[7];
    const float* rbf_W2 = (const float*)d_in[8];
    const float* rbf_b2 = (const float*)d_in[9];
    const float* Wq  = (const float*)d_in[10]; const float* bq  = (const float*)d_in[11];
    const float* Wk  = (const float*)d_in[12]; const float* bk  = (const float*)d_in[13];
    const float* Wv  = (const float*)d_in[14]; const float* bv  = (const float*)d_in[15];
    const float* We  = (const float*)d_in[16]; const float* be  = (const float*)d_in[17];
    const float* Wmu = (const float*)d_in[18]; const float* bmu = (const float*)d_in[19];
    const float* ln1g = (const float*)d_in[20]; const float* ln1b = (const float*)d_in[21];
    const float* Wmsg = (const float*)d_in[22]; const float* bmsg = (const float*)d_in[23];
    const float* ln2g = (const float*)d_in[24]; const float* ln2b = (const float*)d_in[25];
    const float* Wc  = (const float*)d_in[26]; const float* bc  = (const float*)d_in[27];
    const float* bng = (const float*)d_in[28]; const float* bnb = (const float*)d_in[29];
    const float* fcW = (const float*)d_in[30]; const float* fcb = (const float*)d_in[31];
    const float* outW = (const float*)d_in[32]; const float* outb = (const float*)d_in[33];
    float* out = (float*)d_out;

    // ---- workspace carve (256B aligned) ----
    size_t off = 0;
    char* base = (char*)d_ws;
    auto carve = [&](size_t bytes) -> void* {
        void* p = base + off;
        off += (bytes + 255) & ~(size_t)255;
        return p;
    };
    h16* embt  = (h16*)carve((size_t)128 * 96 * 2);
    h16* rw1t  = (h16*)carve((size_t)128 * 128 * 2);
    h16* rw2t  = (h16*)carve((size_t)128 * 128 * 2);
    h16* wqt   = (h16*)carve((size_t)NL_ * 512 * 128 * 2);
    h16* wkt   = (h16*)carve((size_t)NL_ * 512 * 128 * 2);
    h16* wvt   = (h16*)carve((size_t)NL_ * 512 * 128 * 2);
    h16* wet   = (h16*)carve((size_t)NL_ * 512 * 128 * 2);
    h16* wmut  = (h16*)carve((size_t)NL_ * 384 * 384 * 2);
    h16* wmsgt = (h16*)carve((size_t)NL_ * 128 * 384 * 2);
    h16* wct   = (h16*)carve((size_t)NL_ * 128 * 512 * 2);
    h16* xh    = (h16*)carve((size_t)Nn * 96 * 2);
    float* nf  = (float*)carve((size_t)Nn * 128 * 4);
    h16* nfh   = (h16*)carve((size_t)Nn * 128 * 2);
    h16* rbfh  = (h16*)carve((size_t)E_ * 128 * 2);
    h16* hidh  = (h16*)carve((size_t)E_ * 128 * 2);
    h16* efh   = (h16*)carve((size_t)E_ * 128 * 2);
    h16* qh    = (h16*)carve((size_t)Nn * 512 * 2);
    h16* kh    = (h16*)carve((size_t)Nn * 512 * 2);
    h16* vh    = (h16*)carve((size_t)Nn * 512 * 2);
    float* agg = (float*)carve((size_t)Nn * 512 * 4);
    h16* aggh  = (h16*)carve((size_t)Nn * 512 * 2);
    float* obuf = (float*)carve((size_t)Nn * 128 * 4);
    h16* ehc   = (h16*)carve((size_t)CH * 512 * 2);
    h16* gate  = (h16*)carve((size_t)CH * H_ * 384 * 2);
    h16* m3h   = (h16*)carve((size_t)CH * H_ * 384 * 2);
    h16* muh   = (h16*)carve((size_t)CH * H_ * 384 * 2);
    float* msgf = (float*)carve((size_t)CH * H_ * 128 * 4);
    float* pooled = (float*)carve((size_t)G_ * 128 * 4);
    float* cnts   = (float*)carve((size_t)G_ * 4);
    (void)ws_size; (void)n_in;

    // ---- weight pre-transposition to f16 [N,K] (L2-resident, reused everywhere) ----
    auto wtr = [&](const float* W, h16* Wt, int K, int N2, int Kp) {
        int elems = N2 * Kp;
        wtr_kernel<<<(elems + 255) / 256, 256, 0, stream>>>(W, Wt, K, N2, Kp);
    };
    wtr(emb_W, embt, 92, 128, 96);
    wtr(rbf_W1, rw1t, 128, 128, 128);
    wtr(rbf_W2, rw2t, 128, 128, 128);
    for (int l = 0; l < NL_; ++l) {
        wtr(Wq  + (size_t)l * 128 * 512, wqt  + (size_t)l * 512 * 128, 128, 512, 128);
        wtr(Wk  + (size_t)l * 128 * 512, wkt  + (size_t)l * 512 * 128, 128, 512, 128);
        wtr(Wv  + (size_t)l * 128 * 512, wvt  + (size_t)l * 512 * 128, 128, 512, 128);
        wtr(We  + (size_t)l * 128 * 512, wet  + (size_t)l * 512 * 128, 128, 512, 128);
        wtr(Wmu + (size_t)l * 384 * 384, wmut + (size_t)l * 384 * 384, 384, 384, 384);
        wtr(Wmsg+ (size_t)l * 384 * 128, wmsgt+ (size_t)l * 128 * 384, 384, 128, 384);
        wtr(Wc  + (size_t)l * 512 * 128, wct  + (size_t)l * 128 * 512, 512, 128, 512);
    }

    // ---- front: atom embedding + RBF edge MLP ----
    cvt_x_kernel<<<(Nn * 96 + 255) / 256, 256, 0, stream>>>(x, xh, Nn);
    GEMM(0, true,  true,  false, xh, embt, emb_b, (const h16*)nullptr, nf, nfh, Nn, 128, 96);
    rbf_kernel<<<E_, 128, 0, stream>>>(eattr, rbfh, E_);
    GEMM(1, false, true,  false, rbfh, rw1t, rbf_b1, (const h16*)nullptr, (float*)nullptr, hidh, E_, 128, 128);
    GEMM(0, false, true,  false, hidh, rw2t, rbf_b2, (const h16*)nullptr, (float*)nullptr, efh,  E_, 128, 128);

    const float inv_sqrt = 0.05103103630798288f;   // 1/sqrt(3*C)

    // ---- 5 transformer layers ----
    for (int l = 0; l < NL_; ++l) {
        const h16* wqtl = wqt + (size_t)l * 512 * 128;
        const h16* wktl = wkt + (size_t)l * 512 * 128;
        const h16* wvtl = wvt + (size_t)l * 512 * 128;
        const h16* wetl = wet + (size_t)l * 512 * 128;
        const h16* wmutl  = wmut  + (size_t)l * 384 * 384;
        const h16* wmsgtl = wmsgt + (size_t)l * 128 * 384;
        const h16* wctl   = wct   + (size_t)l * 128 * 512;
        const float* bql = bq + (size_t)l * 512, *bkl = bk + (size_t)l * 512;
        const float* bvl = bv + (size_t)l * 512, *bel = be + (size_t)l * 512;
        const float* bmul = bmu + (size_t)l * 384;
        const float* l1g = ln1g + (size_t)l * 384, *l1b = ln1b + (size_t)l * 384;
        const float* bmsgl = bmsg + (size_t)l * 128;
        const float* l2g = ln2g + (size_t)l * 128, *l2b = ln2b + (size_t)l * 128;
        const float* bcl = bc + (size_t)l * 128;
        const float* bngl = bng + (size_t)l * 128, *bnbl = bnb + (size_t)l * 128;

        GEMM(0, false, true, false, nfh, wqtl, bql, (const h16*)nullptr, (float*)nullptr, qh, Nn, 512, 128);
        GEMM(0, false, true, false, nfh, wktl, bkl, (const h16*)nullptr, (float*)nullptr, kh, Nn, 512, 128);
        GEMM(0, false, true, false, nfh, wvtl, bvl, (const h16*)nullptr, (float*)nullptr, vh, Nn, 512, 128);

        long long aggN = (long long)Nn * 512;
        fill0_kernel<<<(int)((aggN + 255) / 256), 256, 0, stream>>>(agg, aggN);

        for (int e0 = 0; e0 < E_; e0 += CH) {
            const int cn = (E_ - e0 < CH) ? (E_ - e0) : CH;
            // e = ef @ We + be  (per-edge, chunk-local, stays in L2)
            GEMM(0, false, true, false, efh + (size_t)e0 * 128, wetl, bel,
                 (const h16*)nullptr, (float*)nullptr, ehc, cn, 512, 128);
            // gather + alpha/LN/sigmoid gate + m3 build
            edge_prep_kernel<<<cn * H_, 384, 0, stream>>>(eidx, E_, e0, qh, kh, vh, ehc,
                                                          l1g, l1b, gate, m3h, inv_sqrt);
            // mu = (m3 @ Wmu + bmu) * gate          [cn*H, 384]
            GEMM(0, false, true, true, m3h, wmutl, bmul, gate, (float*)nullptr, muh,
                 cn * H_, 384, 384);
            // msg = mu @ Wmsg + bmsg                [cn*H, 128] f32
            GEMM(0, true, false, false, muh, wmsgtl, bmsgl, (const h16*)nullptr, msgf,
                 (h16*)nullptr, cn * H_, 128, 384);
            // LN(128) + segment-sum into agg[dst]
            msg_scatter_kernel<<<cn * H_, 128, 0, stream>>>(msgf, eidx, E_, e0, l2g, l2b, agg);
        }

        cvt_f32_f16_kernel<<<(int)((aggN + 255) / 256), 256, 0, stream>>>(agg, aggh, aggN);
        GEMM(0, true, false, false, aggh, wctl, bcl, (const h16*)nullptr, obuf,
             (h16*)nullptr, Nn, 128, 512);
        residual_kernel<<<Nn, 128, 0, stream>>>(obuf, bngl, bnbl, nf, nfh);
    }

    // ---- pooling + head ----
    fill0_kernel<<<(G_ * 128 + 255) / 256, 256, 0, stream>>>(pooled, (long long)G_ * 128);
    fill0_kernel<<<(G_ + 255) / 256, 256, 0, stream>>>(cnts, (long long)G_);
    pool_sum_kernel<<<Nn, 128, 0, stream>>>(nf, batch, pooled, cnts);
    head_kernel<<<G_, 128, 0, stream>>>(pooled, cnts, fcW, fcb, outW, outb, out);
}
